// MultiLatentAttention_2138893714154
// MI455X (gfx1250) — matrix operand order repaired and checked by reading, not compiled, not run on hardware
//
#include <hip/hip_runtime.h>
#include <hip/hip_bf16.h>

// ---------------------------------------------------------------- types
typedef __attribute__((ext_vector_type(16))) __bf16 v16bf;
typedef __attribute__((ext_vector_type(8)))  __bf16 v8bf;
typedef __attribute__((ext_vector_type(8)))  float  v8f;

#define Bc   2
#define Tc   2048
#define Ec   1024
#define Hc   16
#define Dc   64
#define Lc   256
#define BTc  (Bc*Tc)      // 4096

// ---------------------------------------------------------------- WMMA fragment loaders
// A fragment: 16x32 bf16 tile (M x K), row-major, leading dim ld (elements).
// Contiguous K order used here: lanes 0-15 -> M=lane, K {0..15}; lanes 16-31 -> K {16..31}.
__device__ inline v16bf load_frag_a(const __bf16* base, int ld, int lane) {
    union { v16bf v; v8bf h[2]; } u;
    const __bf16* p = base + (size_t)(lane & 15) * ld + ((lane >> 4) << 4);
    u.h[0] = *(const v8bf*)p;
    u.h[1] = *(const v8bf*)(p + 8);
    return u.v;
}
// B fragment: 32x16 bf16 tile (K x N) sourced from a *transposed* buffer Bt[N][K]
// (per lane: column N = lane&15, 16 contiguous K starting at (lane>>4)*16).
__device__ inline v16bf load_frag_b(const __bf16* base, int ld, int lane) {
    union { v16bf v; v8bf h[2]; } u;
    const __bf16* p = base + (size_t)(lane & 15) * ld + ((lane >> 4) << 4);
    u.h[0] = *(const v8bf*)p;
    u.h[1] = *(const v8bf*)(p + 8);
    return u.v;
}
// B fragment out of an LDS tile Bl[64][32] (rows = N, contiguous K).
__device__ inline v16bf load_frag_b_lds(const __bf16 (*Bl)[32], int j, int lane) {
    union { v16bf v; __bf16 e[16]; } u;
    const int row = 16 * j + (lane & 15);
    const int c0 = (lane >> 4) * 16;
#pragma unroll
    for (int i = 0; i < 16; ++i) u.e[i] = Bl[row][c0 + i];
    return u.v;
}

// CDNA5 async global -> LDS copy (16B per lane), tracked by ASYNCcnt.
__device__ inline void async_copy_b128(unsigned ldsoff, const void* gaddr) {
    asm volatile("global_load_async_to_lds_b128 %0, %1, off"
                 :: "v"(ldsoff), "v"(gaddr) : "memory");
}
__device__ inline void wait_async0() {
    asm volatile("s_wait_asynccnt 0" ::: "memory");
}

// ---------------------------------------------------------------- GEMM: C[M][N](f32) = A[M][K](bf16) @ Wt[N][K](bf16)^T
// block = 128 threads = 4 waves; block tile 128(M) x 64(N); wave tile 32 x 64.
// B tile (64 x 32, shared by all waves) staged in LDS via async copies, double buffered.
__global__ __launch_bounds__(128) void gemm_bf16_kernel(const __bf16* __restrict__ A,
                                                        const __bf16* __restrict__ Bt,
                                                        float* __restrict__ C,
                                                        int M, int N, int K) {
    __shared__ __align__(16) __bf16 Blds[2][64][32];   // 2 x 4KB
    const int tid  = threadIdx.x;
    const int lane = tid & 31;
    const int wave = tid >> 5;
    const int m0 = blockIdx.y * 128 + wave * 32;
    const int n0 = blockIdx.x * 64;

    // async B-tile loader: 128 lanes cover 64 rows x 2 x 16B chunks
    const int brow   = tid >> 1;
    const int bchunk = (tid & 1) << 3;                 // element offset (8 bf16 = 16B)
    const __bf16* bsrc = Bt + (size_t)(n0 + brow) * K + bchunk;
    const unsigned ldsrow = (unsigned)(size_t)(&Blds[0][brow][bchunk]);

    // prologue: stage k=0 tile into buffer 0
    async_copy_b128(ldsrow, bsrc);

    v8f acc[2][4] = {};
    int cur = 0;
    for (int k = 0; k < K; k += 32) {
        wait_async0();
        __syncthreads();                               // tile 'cur' visible to all waves
        if (k + 32 < K)
            async_copy_b128(ldsrow + (unsigned)((cur ^ 1) * 4096), bsrc + k + 32);

        v16bf a0 = load_frag_a(A + (size_t)m0 * K + k, K, lane);
        v16bf a1 = load_frag_a(A + (size_t)(m0 + 16) * K + k, K, lane);
#pragma unroll
        for (int j = 0; j < 4; ++j) {
            v16bf b = load_frag_b_lds(Blds[cur], j, lane);
            acc[0][j] = __builtin_amdgcn_wmma_f32_16x16x32_bf16(false, a0, false, b,
                                                                (short)0, acc[0][j], false, false);
            acc[1][j] = __builtin_amdgcn_wmma_f32_16x16x32_bf16(false, a1, false, b,
                                                                (short)0, acc[1][j], false, false);
        }
        cur ^= 1;
    }

    const int lanen = lane & 15;
    const int mbase = (lane >> 4) * 8;
#pragma unroll
    for (int s = 0; s < 2; ++s)
#pragma unroll
        for (int j = 0; j < 4; ++j)
#pragma unroll
            for (int r = 0; r < 8; ++r)
                C[(size_t)(m0 + 16 * s + mbase + r) * N + n0 + 16 * j + lanen] = acc[s][j][r];
}

// ---------------------------------------------------------------- elementwise helpers
__global__ void cvt_bf16_kernel(const float* __restrict__ src, __bf16* __restrict__ dst, int n) {
    int i = blockIdx.x * blockDim.x + threadIdx.x;
    if (i < n) dst[i] = (__bf16)src[i];
}
// Wt[n][k] = (bf16) W[k][n]   (W is K x N row-major)
__global__ void wtrans_bf16_kernel(const float* __restrict__ W, __bf16* __restrict__ Wt, int K, int N) {
    int i = blockIdx.x * blockDim.x + threadIdx.x;
    if (i >= K * N) return;
    int n = i / K, k = i - n * K;
    Wt[i] = (__bf16)W[(size_t)k * N + n];
}

// RoPE (reference: non-interleaved concat of [x1*cos - x2*sin, x1*sin + x2*cos])
__device__ inline float rope_val(const float* rowbase, int j, int t) {
    int jj = j & 31;
    float x1 = rowbase[2 * jj];
    float x2 = rowbase[2 * jj + 1];
    float f = __expf(-0.28782313662425574f * (float)jj);   // 10000^(-jj/32)
    float th = (float)t * f;
    float sn = __sinf(th), cs = __cosf(th);
    return (j < 32) ? (x1 * cs - x2 * sn) : (x1 * sn + x2 * cs);
}

// q[(b,h,t,0:128)] = scale * [Qc | rope(Qr)]
__global__ void pack_q_kernel(const float* __restrict__ Qc, const float* __restrict__ Qr,
                              __bf16* __restrict__ q) {
    int idx = blockIdx.x * blockDim.x + threadIdx.x;
    if (idx >= Bc * Hc * Tc * 128) return;
    int d = idx & 127;
    int t = (idx >> 7) & (Tc - 1);
    int h = (idx >> 18) & (Hc - 1);
    int b = idx >> 22;
    size_t base = ((size_t)(b * Tc + t) << 10) + (h << 6);
    float val = (d < 64) ? Qc[base + d] : rope_val(Qr + base, d - 64, t);
    q[idx] = (__bf16)(val * 0.088388347648318447f);   // 1/sqrt(2*D)
}
__global__ void pack_k_kernel(const float* __restrict__ Kc, const float* __restrict__ Kr,
                              __bf16* __restrict__ k) {
    int idx = blockIdx.x * blockDim.x + threadIdx.x;
    if (idx >= Bc * Hc * Tc * 128) return;
    int d = idx & 127;
    int t = (idx >> 7) & (Tc - 1);
    int h = (idx >> 18) & (Hc - 1);
    int b = idx >> 22;
    size_t base = ((size_t)(b * Tc + t) << 10) + (h << 6);
    float val = (d < 64) ? Kc[base + d] : rope_val(Kr + base, d - 64, t);
    k[idx] = (__bf16)val;
}
// vt[(b,h,d,t)] = V[(b,t,h*64+d)]   (transposed so PV B-fragments are contiguous)
__global__ void pack_v_kernel(const float* __restrict__ Vf, __bf16* __restrict__ vt) {
    int idx = blockIdx.x * blockDim.x + threadIdx.x;
    if (idx >= Bc * Hc * Dc * Tc) return;
    int t = idx & (Tc - 1);
    int d = (idx >> 11) & 63;
    int bh = idx >> 17;
    int b = bh >> 4, h = bh & 15;
    vt[idx] = (__bf16)Vf[((size_t)(b * Tc + t) << 10) + (h << 6) + d];
}
// y_bf16[(b,t,h*64+d)] = obuf[(b,h,t,d)]
__global__ void unpack_o_kernel(const float* __restrict__ obuf, __bf16* __restrict__ y) {
    int idx = blockIdx.x * blockDim.x + threadIdx.x;
    if (idx >= Bc * Tc * Ec) return;
    int e = idx & (Ec - 1);
    int t = (idx >> 10) & (Tc - 1);
    int b = idx >> 21;
    int h = e >> 6, d = e & 63;
    y[idx] = (__bf16)obuf[(((size_t)(b * Hc + h) * Tc) + t) * 64 + d];
}

// ---------------------------------------------------------------- flash attention (causal)
// grid: (T/64, B*H); block: 128 (4 waves). Wave owns 16 query rows, streams keys in 32-chunks.
__global__ __launch_bounds__(128) void flash_kernel(const __bf16* __restrict__ qb,
                                                    const __bf16* __restrict__ kb_,
                                                    const __bf16* __restrict__ vt,
                                                    float* __restrict__ obuf) {
    const int lane = threadIdx.x & 31;
    const int wave = threadIdx.x >> 5;
    const int bh = blockIdx.y;
    const int q0 = blockIdx.x * 64 + wave * 16;
    const __bf16* Q  = qb  + (size_t)bh * Tc * 128;
    const __bf16* Kp = kb_ + (size_t)bh * Tc * 128;
    const __bf16* Vp = vt  + (size_t)bh * Dc * Tc;

    __shared__ __align__(16) __bf16 Plds[4][16][32];

    const int lanen = lane & 15;
    const int mbase = (lane >> 4) * 8;

    v16bf qa[4];
#pragma unroll
    for (int c = 0; c < 4; ++c) qa[c] = load_frag_a(Q + (size_t)q0 * 128 + 32 * c, 128, lane);

    float mrow[8], lrow[8];
    v8f acc[4] = {};
#pragma unroll
    for (int r = 0; r < 8; ++r) { mrow[r] = -1e30f; lrow[r] = 0.0f; }

    const int kend = q0 + 16;   // causal: keys < q0+16 needed by this wave
    for (int kb = 0; kb < kend; kb += 32) {
        // S = Q @ K^T over d=128, two 16-key tiles
        v8f s0 = {}, s1 = {};
#pragma unroll
        for (int c = 0; c < 4; ++c) {
            v16bf b0 = load_frag_b(Kp + (size_t)kb * 128 + 32 * c, 128, lane);
            s0 = __builtin_amdgcn_wmma_f32_16x16x32_bf16(false, qa[c], false, b0,
                                                         (short)0, s0, false, false);
            v16bf b1 = load_frag_b(Kp + (size_t)(kb + 16) * 128 + 32 * c, 128, lane);
            s1 = __builtin_amdgcn_wmma_f32_16x16x32_bf16(false, qa[c], false, b1,
                                                         (short)0, s1, false, false);
        }
        // online softmax (row = VGPR slot; reduce over 16-lane half-wave = N dim)
        float scl[8];
#pragma unroll
        for (int r = 0; r < 8; ++r) {
            int qi = q0 + mbase + r;
            float a0 = (kb + lanen      <= qi) ? s0[r] : -1e30f;
            float a1 = (kb + 16 + lanen <= qi) ? s1[r] : -1e30f;
            float mx = fmaxf(a0, a1);
#pragma unroll
            for (int msk = 1; msk < 16; msk <<= 1) mx = fmaxf(mx, __shfl_xor(mx, msk, 32));
            float mn = fmaxf(mrow[r], mx);
            float p0 = __expf(a0 - mn);
            float p1 = __expf(a1 - mn);
            float sum = p0 + p1;
#pragma unroll
            for (int msk = 1; msk < 16; msk <<= 1) sum += __shfl_xor(sum, msk, 32);
            scl[r]  = __expf(mrow[r] - mn);
            lrow[r] = lrow[r] * scl[r] + sum;
            mrow[r] = mn;
            Plds[wave][mbase + r][lanen]      = (__bf16)p0;
            Plds[wave][mbase + r][16 + lanen] = (__bf16)p1;
        }
#pragma unroll
        for (int c = 0; c < 4; ++c)
#pragma unroll
            for (int r = 0; r < 8; ++r) acc[c][r] *= scl[r];

        // order cross-lane LDS write -> read (per-wave region, DS in-order + explicit wait)
        asm volatile("s_wait_dscnt 0" ::: "memory");

        // P fragment from LDS (A layout)
        union { v16bf v; __bf16 e[16]; } pa;
        {
            int c0 = (lane >> 4) * 16;
#pragma unroll
            for (int i = 0; i < 8; ++i) {
                pa.e[i]     = Plds[wave][lanen][c0 + i];
                pa.e[8 + i] = Plds[wave][lanen][c0 + 8 + i];
            }
        }
        // O += P @ V  (V transposed: Vt[d][t], contiguous in t)
#pragma unroll
        for (int c = 0; c < 4; ++c) {
            v16bf bv = load_frag_b(Vp + (size_t)(c * 16) * Tc + kb, Tc, lane);
            acc[c] = __builtin_amdgcn_wmma_f32_16x16x32_bf16(false, pa.v, false, bv,
                                                             (short)0, acc[c], false, false);
        }
    }
    float* O = obuf + (size_t)bh * Tc * 64;
#pragma unroll
    for (int c = 0; c < 4; ++c)
#pragma unroll
        for (int r = 0; r < 8; ++r) {
            int qi = q0 + mbase + r;
            O[(size_t)qi * 64 + c * 16 + lanen] = acc[c][r] / lrow[r];
        }
}

// ---------------------------------------------------------------- host orchestration
extern "C" void kernel_launch(void* const* d_in, const int* in_sizes, int n_in,
                              void* d_out, int out_size, void* d_ws, size_t ws_size,
                              hipStream_t stream) {
    const float* x    = (const float*)d_in[0];
    const float* Wdq  = (const float*)d_in[1];
    const float* Wuq  = (const float*)d_in[2];
    const float* Wqr  = (const float*)d_in[3];
    const float* Wdkv = (const float*)d_in[4];
    const float* Wuk  = (const float*)d_in[5];
    const float* Wkr  = (const float*)d_in[6];
    const float* Wuv  = (const float*)d_in[7];
    const float* Wo   = (const float*)d_in[8];
    float* out = (float*)d_out;

    size_t off = 0;
    auto alloc = [&](size_t bytes) -> void* {
        void* p = (char*)d_ws + off;
        off += (bytes + 255) & ~(size_t)255;
        return p;
    };
    const size_t nBTE = (size_t)BTc * Ec;           // 4 Mi elems
    const size_t nBTL = (size_t)BTc * Lc;           // 1 Mi elems

    __bf16* xbf    = (__bf16*)alloc(nBTE * 2);
    __bf16* wdq_t  = (__bf16*)alloc((size_t)Ec * Ec * 2);
    __bf16* wuq_t  = (__bf16*)alloc((size_t)Ec * Ec * 2);
    __bf16* wqr_t  = (__bf16*)alloc((size_t)Ec * Ec * 2);
    __bf16* wkr_t  = (__bf16*)alloc((size_t)Ec * Ec * 2);
    __bf16* wo_t   = (__bf16*)alloc((size_t)Ec * Ec * 2);
    __bf16* wdkv_t = (__bf16*)alloc((size_t)Lc * Ec * 2);   // [N=L][K=E]
    __bf16* wuk_t  = (__bf16*)alloc((size_t)Ec * Lc * 2);   // [N=E][K=L]
    __bf16* wuv_t  = (__bf16*)alloc((size_t)Ec * Lc * 2);
    float*  Cq     = (float*)alloc(nBTE * 4);
    __bf16* Cqbf   = (__bf16*)alloc(nBTE * 2);
    float*  Ckv    = (float*)alloc(nBTL * 4);
    __bf16* Ckvbf  = (__bf16*)alloc(nBTL * 2);
    float*  Qc     = (float*)alloc(nBTE * 4);
    float*  Qr     = (float*)alloc(nBTE * 4);
    float*  Kc     = (float*)alloc(nBTE * 4);
    float*  Krr    = (float*)alloc(nBTE * 4);
    float*  Vf     = (float*)alloc(nBTE * 4);
    __bf16* qbf    = (__bf16*)alloc((size_t)Bc * Hc * Tc * 128 * 2);
    __bf16* kbf    = (__bf16*)alloc((size_t)Bc * Hc * Tc * 128 * 2);
    __bf16* vtbf   = (__bf16*)alloc((size_t)Bc * Hc * Dc * Tc * 2);
    float*  obuf   = (float*)alloc((size_t)Bc * Hc * Tc * Dc * 4);
    __bf16* ybf    = (__bf16*)alloc(nBTE * 2);
    (void)ws_size; (void)in_sizes; (void)n_in; (void)out_size;

    const int TPB = 256;
    auto nb = [](size_t n) { return (unsigned)((n + 255) / 256); };

    // convert activations & weights
    cvt_bf16_kernel<<<nb(nBTE), TPB, 0, stream>>>(x, xbf, (int)nBTE);
    wtrans_bf16_kernel<<<nb((size_t)Ec * Ec), TPB, 0, stream>>>(Wdq,  wdq_t,  Ec, Ec);
    wtrans_bf16_kernel<<<nb((size_t)Ec * Ec), TPB, 0, stream>>>(Wuq,  wuq_t,  Ec, Ec);
    wtrans_bf16_kernel<<<nb((size_t)Ec * Ec), TPB, 0, stream>>>(Wqr,  wqr_t,  Ec, Ec);
    wtrans_bf16_kernel<<<nb((size_t)Ec * Ec), TPB, 0, stream>>>(Wkr,  wkr_t,  Ec, Ec);
    wtrans_bf16_kernel<<<nb((size_t)Ec * Ec), TPB, 0, stream>>>(Wo,   wo_t,   Ec, Ec);
    wtrans_bf16_kernel<<<nb((size_t)Ec * Lc), TPB, 0, stream>>>(Wdkv, wdkv_t, Ec, Lc);
    wtrans_bf16_kernel<<<nb((size_t)Lc * Ec), TPB, 0, stream>>>(Wuk,  wuk_t,  Lc, Ec);
    wtrans_bf16_kernel<<<nb((size_t)Lc * Ec), TPB, 0, stream>>>(Wuv,  wuv_t,  Lc, Ec);

    dim3 blk(128);
    // Cq = x @ Wdq
    gemm_bf16_kernel<<<dim3(Ec / 64, BTc / 128), blk, 0, stream>>>(xbf, wdq_t, Cq, BTc, Ec, Ec);
    cvt_bf16_kernel<<<nb(nBTE), TPB, 0, stream>>>(Cq, Cqbf, (int)nBTE);
    // Qc = Cq @ Wuq ; Qr = Cq @ Wqr
    gemm_bf16_kernel<<<dim3(Ec / 64, BTc / 128), blk, 0, stream>>>(Cqbf, wuq_t, Qc, BTc, Ec, Ec);
    gemm_bf16_kernel<<<dim3(Ec / 64, BTc / 128), blk, 0, stream>>>(Cqbf, wqr_t, Qr, BTc, Ec, Ec);
    // Ckv = x @ Wdkv
    gemm_bf16_kernel<<<dim3(Lc / 64, BTc / 128), blk, 0, stream>>>(xbf, wdkv_t, Ckv, BTc, Lc, Ec);
    cvt_bf16_kernel<<<nb(nBTL), TPB, 0, stream>>>(Ckv, Ckvbf, (int)nBTL);
    // Kc = Ckv @ Wuk ; Kr = x @ Wkr ; V = Ckv @ Wuv
    gemm_bf16_kernel<<<dim3(Ec / 64, BTc / 128), blk, 0, stream>>>(Ckvbf, wuk_t, Kc, BTc, Ec, Lc);
    gemm_bf16_kernel<<<dim3(Ec / 64, BTc / 128), blk, 0, stream>>>(xbf, wkr_t, Krr, BTc, Ec, Ec);
    gemm_bf16_kernel<<<dim3(Ec / 64, BTc / 128), blk, 0, stream>>>(Ckvbf, wuv_t, Vf, BTc, Ec, Lc);

    // pack q/k (RoPE + scale) and v (transposed)
    pack_q_kernel<<<nb((size_t)Bc * Hc * Tc * 128), TPB, 0, stream>>>(Qc, Qr, qbf);
    pack_k_kernel<<<nb((size_t)Bc * Hc * Tc * 128), TPB, 0, stream>>>(Kc, Krr, kbf);
    pack_v_kernel<<<nb((size_t)Bc * Hc * Dc * Tc), TPB, 0, stream>>>(Vf, vtbf);

    // causal flash attention
    flash_kernel<<<dim3(Tc / 64, Bc * Hc), blk, 0, stream>>>(qbf, kbf, vtbf, obuf);

    // repack heads and output projection
    unpack_o_kernel<<<nb(nBTE), TPB, 0, stream>>>(obuf, ybf);
    gemm_bf16_kernel<<<dim3(Ec / 64, BTc / 128), blk, 0, stream>>>(ybf, wo_t, out, BTc, Ec, Ec);
}
